// adAttention_64639257804907
// MI455X (gfx1250) — matrix operand order repaired and checked by reading, not compiled, not run on hardware
//
#include <hip/hip_runtime.h>
#include <hip/hip_bf16.h>
#include <math.h>

// Shapes (fixed by the reference)
#define B_ 4
#define L_ 2048
#define H_ 8
#define E_ 64
#define D_ 64
#define EPS_ 1e-6f
// scale folded into q_adj: (1/sqrt(64)) * log2(e), so inner loop uses exp2 directly
#define QK_SCALE_ 0.1803368801111204f

typedef __attribute__((ext_vector_type(16))) _Float16 v16h;
typedef __attribute__((ext_vector_type(8)))  _Float16 h8;
typedef __attribute__((ext_vector_type(8)))  float    v8f;
typedef __attribute__((__vector_size__(4 * sizeof(int)))) int v4i;

typedef v4i __attribute__((address_space(1))) global_v4i;
typedef v4i __attribute__((address_space(3))) shared_v4i;

#define HAS_ASYNC_LDS __has_builtin(__builtin_amdgcn_global_load_async_to_lds_b128)

// ---------------------------------------------------------------------------
// async helpers (CDNA5 ASYNCcnt path), with safe fallbacks
// ---------------------------------------------------------------------------
__device__ __forceinline__ void wait_async_le(int n) {
#if __has_builtin(__builtin_amdgcn_s_wait_asynccnt)
    if (n) __builtin_amdgcn_s_wait_asynccnt(4);
    else   __builtin_amdgcn_s_wait_asynccnt(0);
#else
    if (n) asm volatile("s_wait_asynccnt 0x4" ::: "memory");
    else   asm volatile("s_wait_asynccnt 0x0" ::: "memory");
#endif
}

#if HAS_ASYNC_LDS
// Stage one 64x64 f16 tile (8KB, contiguous) global -> LDS via async copies
// tracked by ASYNCcnt: 128 threads x 4 x b128.
__device__ __forceinline__ void stage_k_async(const _Float16* __restrict__ src,
                                              _Float16* dst, int tid) {
#pragma unroll
    for (int i = 0; i < 4; ++i) {
        int idx = (tid + i * 128) * 8;   // halfs; 16B per op
        __builtin_amdgcn_global_load_async_to_lds_b128(
            (global_v4i*)(v4i*)(src + idx),
            (shared_v4i*)(v4i*)(dst + idx),
            0, 0);
    }
}
#endif

// ---------------------------------------------------------------------------
// Pass 1: column sums over L for q and k:  qsum/ksum [B*H, 64]
// ---------------------------------------------------------------------------
__global__ void sums_kernel(const float* __restrict__ q, const float* __restrict__ k,
                            float* __restrict__ qsum, float* __restrict__ ksum) {
    int bh = blockIdx.x;
    int b = bh / H_, h = bh % H_;
    int e = threadIdx.x;
    float sq = 0.f, sk = 0.f;
    size_t base = ((size_t)b * L_ * H_ + h) * E_ + e;
    const size_t stride = (size_t)H_ * E_;
    for (int l = 0; l < L_; ++l) { sq += q[base]; sk += k[base]; base += stride; }
    qsum[bh * E_ + e] = sq;
    ksum[bh * E_ + e] = sk;
}

// ---------------------------------------------------------------------------
// Pass 2: adjust (signed-sqrt + L2 norm) + cast to f16, relayout to [B,H,L,64]
// q_adj additionally carries the softmax scale in log2 domain.
// ---------------------------------------------------------------------------
__global__ void adjust_kernel(const float* __restrict__ q, const float* __restrict__ k,
                              const float* __restrict__ v,
                              const float* __restrict__ qsum, const float* __restrict__ ksum,
                              const float* __restrict__ alpha,
                              _Float16* __restrict__ qadj, _Float16* __restrict__ kadj,
                              _Float16* __restrict__ vh) {
    int l  = blockIdx.x;
    int bh = blockIdx.y;
    int b = bh / H_, h = bh % H_;
    int e = threadIdx.x;
    float al = alpha[0];
    size_t in_off  = (((size_t)b * L_ + l) * H_ + h) * E_ + e;
    size_t out_off = ((size_t)bh * L_ + l) * (size_t)E_ + e;
    __shared__ float red[E_];

    // value cast (bandwidth-trivial)
    vh[out_off] = (_Float16)v[in_off];

    // q_adj (scale folded in)
    float x = q[in_off] + al * ksum[bh * E_ + e];
    float y = copysignf(sqrtf(fabsf(x) + EPS_), x);
    red[e] = y * y; __syncthreads();
    for (int off = E_ / 2; off > 0; off >>= 1) { if (e < off) red[e] += red[e + off]; __syncthreads(); }
    float inv = QK_SCALE_ / (sqrtf(red[0]) + EPS_);
    qadj[out_off] = (_Float16)(y * inv);
    __syncthreads();

    // k_adj
    x = k[in_off] + al * qsum[bh * E_ + e];
    y = copysignf(sqrtf(fabsf(x) + EPS_), x);
    red[e] = y * y; __syncthreads();
    for (int off = E_ / 2; off > 0; off >>= 1) { if (e < off) red[e] += red[e + off]; __syncthreads(); }
    inv = 1.0f / (sqrtf(red[0]) + EPS_);
    kadj[out_off] = (_Float16)(y * inv);
}

// ---------------------------------------------------------------------------
// Pass 3: flash-attention forward, WMMA f32_16x16x32_f16, wave32.
// Block = 128 threads (4 waves). 64 M-rows per block, 16 per wave.
// K tiles streamed via GLOBAL_LOAD_ASYNC_TO_LDS (double-buffered); V tiles
// prefetched into registers and transposed into LDS for B-fragment reads.
// ---------------------------------------------------------------------------
__global__ __launch_bounds__(128) void flash_kernel(const _Float16* __restrict__ qa,
                                                    const _Float16* __restrict__ ka,
                                                    const _Float16* __restrict__ vh,
                                                    float* __restrict__ out) {
    __shared__ _Float16 ldsK [2][64 * 64];   // K tile, row-major [n][e], double-buffered
    __shared__ _Float16 ldsVt[2][64 * 64];   // V tile, transposed [d][n], double-buffered
    __shared__ _Float16 ldsP [4 * 16 * 64];  // per-wave P tiles [m][n]

    const int tid   = threadIdx.x;
    const int wave  = tid >> 5;
    const int lane  = tid & 31;
    const int lhalf = lane >> 4;
    const int l16   = lane & 15;

    const int m0 = blockIdx.x * 64;
    const int bh = blockIdx.y;
    const int b = bh / H_, h = bh % H_;

    const size_t bh_off = (size_t)bh * L_ * 64;
    const _Float16* kbase = ka + bh_off;
    const _Float16* vbase = vh + bh_off;

    // Q A-fragments (loaded once): row = m0 + wave*16 + l16
    // A 16x32 layout (contiguous, matches B): elem i -> k = lhalf*16+i
    const _Float16* qrow = qa + bh_off + (size_t)(m0 + wave * 16 + l16) * 64;
    v16h aq[2];
#pragma unroll
    for (int f = 0; f < 2; ++f) {
        h8 lo = *(const h8*)(qrow + f * 32 + lhalf * 16);
        h8 hi = *(const h8*)(qrow + f * 32 + lhalf * 16 + 8);
        v16h a{};
#pragma unroll
        for (int i = 0; i < 8; ++i) { a[i] = lo[i]; a[i + 8] = hi[i]; }
        aq[f] = a;
    }

    v8f o[4];
#pragma unroll
    for (int c = 0; c < 4; ++c) o[c] = v8f{};
    float m8[8], l8[8];
#pragma unroll
    for (int r = 0; r < 8; ++r) { m8[r] = -3.0e38f; l8[r] = 0.f; }

    const int jmax = m0 >> 6;   // causal: only key blocks with n0 <= m0

    // ---- preamble: start tile 0 in flight ----
#if HAS_ASYNC_LDS
    stage_k_async(kbase, &ldsK[0][0], tid);
#endif
    h8 vreg[4];
#pragma unroll
    for (int i = 0; i < 4; ++i) vreg[i] = ((const h8*)vbase)[tid + i * 128];

    for (int j = 0; j <= jmax; ++j) {
        const int n0 = j * 64;
        const int buf = j & 1;
        __syncthreads();   // all waves done reading buffers from iteration j-1

        // kick off K tile j+1 (async copy into the other buffer)
#if HAS_ASYNC_LDS
        if (j < jmax) stage_k_async(kbase + (size_t)(n0 + 64) * 64, &ldsK[buf ^ 1][0], tid);
#else
        { // synchronous fallback: stage K tile j now
            const h8* kt = (const h8*)(kbase + (size_t)n0 * 64);
#pragma unroll
            for (int i = 0; i < 4; ++i) {
                int idx = tid + i * 128;
                ((h8*)&ldsK[buf][0])[idx] = kt[idx];
            }
        }
#endif

        // transpose-store prefetched V tile j into LDS [d][n]
#pragma unroll
        for (int i = 0; i < 4; ++i) {
            int idx = tid + i * 128;
            int n = idx >> 3, d0 = (idx & 7) * 8;
#pragma unroll
            for (int u = 0; u < 8; ++u) ldsVt[buf][(d0 + u) * 64 + n] = vreg[i][u];
        }
        // prefetch V tile j+1 into registers
        if (j < jmax) {
            const h8* vt = (const h8*)(vbase + (size_t)(n0 + 64) * 64);
#pragma unroll
            for (int i = 0; i < 4; ++i) vreg[i] = vt[tid + i * 128];
        }

#if HAS_ASYNC_LDS
        wait_async_le(j < jmax ? 4 : 0);   // tile j's async copies complete (in-order)
#endif
        __syncthreads();   // K tile j + V transpose visible to all waves

        const _Float16* Kj = &ldsK[buf][0];
        const _Float16* Vj = &ldsVt[buf][0];

        // ---- S = Qadj · Kadj^T (16x64 per wave), already in log2 domain ----
        v8f s[4];
#pragma unroll
        for (int c = 0; c < 4; ++c) {
            v8f acc{};
#pragma unroll
            for (int ks = 0; ks < 2; ++ks) {
                const v16h bf = *(const v16h*)(Kj + (c * 16 + l16) * 64 + ks * 32 + lhalf * 16);
                acc = __builtin_amdgcn_wmma_f32_16x16x32_f16(false, aq[ks], false, bf,
                                                             (short)0, acc, false, false);
            }
            s[c] = acc;
        }

        // ---- causal mask on diagonal block ----
        if (j == jmax) {
#pragma unroll
            for (int c = 0; c < 4; ++c) {
#pragma unroll
                for (int r = 0; r < 8; ++r) {
                    int row = m0 + wave * 16 + r + 8 * lhalf; // C layout: M = r + 8*(lane/16)
                    int col = n0 + c * 16 + l16;
                    if (col > row) s[c][r] = -3.0e38f;
                }
            }
        }

        // ---- online softmax (base-2; row stats reduced over 16-lane half) ----
        float corr[8];
#pragma unroll
        for (int r = 0; r < 8; ++r) {
            float rmax = s[0][r];
#pragma unroll
            for (int c = 1; c < 4; ++c) rmax = fmaxf(rmax, s[c][r]);
#pragma unroll
            for (int off = 1; off < 16; off <<= 1)
                rmax = fmaxf(rmax, __shfl_xor(rmax, off, 32));
            float mn = fmaxf(m8[r], rmax);
            corr[r] = exp2f(m8[r] - mn);
            m8[r] = mn;
        }
        float rsum[8];
#pragma unroll
        for (int r = 0; r < 8; ++r) rsum[r] = 0.f;
#pragma unroll
        for (int c = 0; c < 4; ++c) {
#pragma unroll
            for (int r = 0; r < 8; ++r) {
                float p = exp2f(s[c][r] - m8[r]);
                s[c][r] = p;
                rsum[r] += p;
            }
        }
#pragma unroll
        for (int r = 0; r < 8; ++r) {
            float t = rsum[r];
#pragma unroll
            for (int off = 1; off < 16; off <<= 1) t += __shfl_xor(t, off, 32);
            l8[r] = l8[r] * corr[r] + t;
        }

        // ---- P -> per-wave LDS (C layout -> row-major), reload as A frags ----
        _Float16* pw = ldsP + wave * 16 * 64;
#pragma unroll
        for (int c = 0; c < 4; ++c) {
#pragma unroll
            for (int r = 0; r < 8; ++r)
                pw[(r + 8 * lhalf) * 64 + c * 16 + l16] = (_Float16)s[c][r];
        }

        // rescale O by correction factor (same C row mapping)
#pragma unroll
        for (int c = 0; c < 4; ++c) {
#pragma unroll
            for (int r = 0; r < 8; ++r) o[c][r] *= corr[r];
        }

        // ---- O += P · V ----
#pragma unroll
        for (int ks = 0; ks < 2; ++ks) {
            h8 lo = *(const h8*)(pw + l16 * 64 + ks * 32 + lhalf * 16);
            h8 hi = *(const h8*)(pw + l16 * 64 + ks * 32 + lhalf * 16 + 8);
            v16h pa{};
#pragma unroll
            for (int i = 0; i < 8; ++i) { pa[i] = lo[i]; pa[i + 8] = hi[i]; }
#pragma unroll
            for (int c = 0; c < 4; ++c) {
                const v16h bv = *(const v16h*)(Vj + (c * 16 + l16) * 64 + ks * 32 + lhalf * 16);
                o[c] = __builtin_amdgcn_wmma_f32_16x16x32_f16(false, pa, false, bv,
                                                              (short)0, o[c], false, false);
            }
        }
    }

    // ---- epilogue: divide by l, store to [B,L,H,D] fp32 ----
#pragma unroll
    for (int c = 0; c < 4; ++c) {
#pragma unroll
        for (int r = 0; r < 8; ++r) {
            int row = m0 + wave * 16 + r + 8 * lhalf;
            int d   = c * 16 + l16;
            out[(((size_t)b * L_ + row) * H_ + h) * D_ + d] = o[c][r] / l8[r];
        }
    }
}

// ---------------------------------------------------------------------------
// Launch: sums -> adjust/cast -> flash attention
// ---------------------------------------------------------------------------
extern "C" void kernel_launch(void* const* d_in, const int* in_sizes, int n_in,
                              void* d_out, int out_size, void* d_ws, size_t ws_size,
                              hipStream_t stream) {
    (void)in_sizes; (void)n_in; (void)out_size; (void)ws_size;
    const float* q     = (const float*)d_in[0];
    const float* k     = (const float*)d_in[1];
    const float* v     = (const float*)d_in[2];
    // d_in[3] = attn_mask (triangular causal) — handled analytically, never read
    const float* alpha = (const float*)d_in[4];
    float* out = (float*)d_out;

    // workspace layout (bytes):
    //   [0, 8K)        qsum  f32 [B*H,64]
    //   [8K, 16K)      ksum  f32 [B*H,64]
    //   [16K, +8M)     qadj  f16 [B,H,L,64]  (scale*log2e folded in)
    //   [+8M, +16M)    kadj  f16 [B,H,L,64]
    //   [+16M, +24M)   vh    f16 [B,H,L,64]
    char* ws = (char*)d_ws;
    float*     qsum = (float*)(ws);
    float*     ksum = (float*)(ws + 8192);
    _Float16*  qadj = (_Float16*)(ws + 16384);
    _Float16*  kadj = (_Float16*)(ws + 16384 + (size_t)8388608);
    _Float16*  vh   = (_Float16*)(ws + 16384 + (size_t)2 * 8388608);

    sums_kernel<<<dim3(B_ * H_), dim3(E_), 0, stream>>>(q, k, qsum, ksum);
    adjust_kernel<<<dim3(L_, B_ * H_), dim3(E_), 0, stream>>>(q, k, v, qsum, ksum, alpha,
                                                              qadj, kadj, vh);
    flash_kernel<<<dim3(L_ / 64, B_ * H_), dim3(128), 0, stream>>>(qadj, kadj, vh, out);
}
